// LossComputation_40733469835975
// MI455X (gfx1250) — compile-verified, matrix operand order checked by reading, not run on hardware
//
#include <hip/hip_runtime.h>
#include <hip/hip_bf16.h>

#define SCALE   28.0f
#define ALPHA   0.6f
#define BETA    0.4f
#define SPOS    10.0f
#define SNEG    40.0f
#define NCLS    11003
#define FEAT    2048
#define BATCH   1024
#define MROWS   2048            /* stacked [v; t] */
#define NPADW   11008           /* 86 * 128 */
#define NBLK    86              /* N blocks of 128 */
#define KTILES  64              /* FEAT / 32 */

typedef __attribute__((ext_vector_type(16))) __bf16 v16bf;
typedef __attribute__((ext_vector_type(8)))  float  v8f;

/* ---------------- workspace layout (BYTE offsets, all 256B-aligned) ----
   rowInv   :          0 +   8,192
   lblLogit :      8,192 +   8,192
   partials :     16,384 + 1,409,024   (2048 rows * 86 blocks * {max,sum})
   rowLoss  :  1,425,408 +   8,192
   simPart  :  1,433,600 +   1,024
   Ahi      :  1,434,624 + 8,388,608   (fragment-major bf16, normalized [v;t])
   Alo      :  9,823,232 + 8,388,608
   Whi      : 18,211,840 + 45,088,768  (fragment-major bf16, normalized W)
   Wlo      : 63,300,608 + 45,088,768
   total ~103.4 MB                                                       */
#define B_ROWINV   0u
#define B_LBL      8192u
#define B_PART     16384u
#define B_ROWLOSS  1425408u
#define B_SIMPART  1433600u
#define B_AHI      1434624u
#define B_ALO      9823232u
#define B_WHI      18211840u
#define B_WLO      63300608u

__device__ __forceinline__ float softplusf(float x) {
    return fmaxf(x, 0.0f) + log1pf(expf(-fabsf(x)));
}

/* fragment-order permutation: chunk slot i -> k offset within 32-wide k-tile.
   slots [0..7]->k 0..7, [8..15]->k 16..23, [16..23]->k 8..15, [24..31]->k 24..31
   so a lane (half = lane>>4) reads its 16-element WMMA fragment as one
   contiguous 32B chunk at (chunk*32 + half*16).                           */
__device__ __forceinline__ int frag_perm(int i) {
    return (i < 8) ? i : (i < 16) ? (i + 8) : (i < 24) ? (i - 8) : i;
}

__device__ __forceinline__ v16bf ldfrag(const __bf16* base, size_t chunk, int half) {
    return *(const v16bf*)(base + chunk * 32 + (size_t)half * 16);
}

__device__ __forceinline__ v8f wmma_bf16(v16bf a, v16bf b, v8f c) {
    return __builtin_amdgcn_wmma_f32_16x16x32_bf16(false, a, false, b, (short)0, c,
                                                   false, false);
}

/* -------- kernel 1: per-row inverse L2 norms of [v; t] -------- */
__global__ __launch_bounds__(256) void rownorm_kernel(const float* __restrict__ vis,
                                                      const float* __restrict__ txt,
                                                      float* __restrict__ rowInv) {
    const int r = blockIdx.x;
    const float* p = (r < BATCH) ? vis + (size_t)r * FEAT
                                 : txt + (size_t)(r - BATCH) * FEAT;
    float s = 0.0f;
    for (int k = threadIdx.x; k < FEAT; k += 256) { float x = p[k]; s += x * x; }
    __shared__ float red[256];
    red[threadIdx.x] = s; __syncthreads();
    for (int off = 128; off > 0; off >>= 1) {
        if (threadIdx.x < off) red[threadIdx.x] += red[threadIdx.x + off];
        __syncthreads();
    }
    if (threadIdx.x == 0) rowInv[r] = rsqrtf(red[0]);
}

/* -------- kernel 2: normalize [v;t] rows, split bf16 hi/lo, fragment-major -- */
__global__ __launch_bounds__(256) void aconv_kernel(const float* __restrict__ vis,
                                                    const float* __restrict__ txt,
                                                    const float* __restrict__ rowInv,
                                                    __bf16* __restrict__ ahi,
                                                    __bf16* __restrict__ alo) {
    const int tid = blockIdx.x * 256 + threadIdx.x;   /* 2048 rows * 64 ktiles */
    const int r   = tid & (MROWS - 1);
    const int kt  = tid >> 11;
    const float* p = (r < BATCH) ? vis + (size_t)r * FEAT
                                 : txt + (size_t)(r - BATCH) * FEAT;
    const float rinv = rowInv[r];
    const size_t off = ((size_t)kt * MROWS + r) * 32;
#pragma unroll
    for (int i = 0; i < 32; ++i) {
        const int k = kt * 32 + frag_perm(i);
        const float x = p[k] * rinv;
        const __bf16 h = (__bf16)x;
        ahi[off + i] = h;
        alo[off + i] = (__bf16)(x - (float)h);
    }
}

/* -------- kernel 3: W column-norm + split bf16 hi/lo, fragment-major --------
   Second pass over W hits L2 (W = 90 MB < 192 MB L2). Padded cols -> zeros. */
__global__ __launch_bounds__(256) void wconv_kernel(const float* __restrict__ W,
                                                    __bf16* __restrict__ whi,
                                                    __bf16* __restrict__ wlo) {
    const int c = blockIdx.x * 256 + threadIdx.x;     /* 43*256 = 11008 */
    if (c >= NPADW) return;
    if (c >= NCLS) {
        for (int kt = 0; kt < KTILES; ++kt) {
            const size_t off = ((size_t)kt * NPADW + c) * 32;
#pragma unroll
            for (int i = 0; i < 32; ++i) { whi[off + i] = (__bf16)0.0f;
                                           wlo[off + i] = (__bf16)0.0f; }
        }
        return;
    }
    float s = 0.0f;
    for (int k = 0; k < FEAT; ++k) { float w = W[(size_t)k * NCLS + c]; s += w * w; }
    const float winv = rsqrtf(s);
    for (int kt = 0; kt < KTILES; ++kt) {
        const size_t off = ((size_t)kt * NPADW + c) * 32;
#pragma unroll
        for (int i = 0; i < 32; ++i) {
            const int k = kt * 32 + frag_perm(i);
            const float x = W[(size_t)k * NCLS + c] * winv;
            const __bf16 h = (__bf16)x;
            whi[off + i] = h;
            wlo[off + i] = (__bf16)(x - (float)h);
        }
    }
}

/* -------- kernel 4: cosine-classifier GEMM + fused partial softmax --------
   grid (86, 32); block tile 64(M) x 128(N); 8 waves = 2(M) x 4(N), each wave
   a 32x32 tile (4 accumulators) for 683 B loaded per WMMA.
   Split-bf16 f32 emulation: Ah*Bh + Ah*Bl + Al*Bh.                          */
__global__ __launch_bounds__(256) void inst_gemm_kernel(
    const __bf16* __restrict__ Ahi, const __bf16* __restrict__ Alo,
    const __bf16* __restrict__ Whi, const __bf16* __restrict__ Wlo,
    const int* __restrict__ labels,
    float* __restrict__ partials, float* __restrict__ lblLogit) {

    __shared__ float tile[64][132];

    const int lane = threadIdx.x & 31;
    const int wave = threadIdx.x >> 5;
    const int wm   = wave >> 2;           /* 0..1 */
    const int wn   = wave & 3;            /* 0..3 */
    const int half = lane >> 4;
    const int lrow = lane & 15;
    const int blockRow = blockIdx.y * 64;
    const int blockCol = blockIdx.x * 128;

    const int c0 = blockCol + wn * 32 + lrow;   /* < 11008 always */
    const int c1 = c0 + 16;
    const int r0 = blockRow + wm * 32 + lrow;
    const int r1 = r0 + 16;

    v8f acc00 = {}; v8f acc01 = {}; v8f acc10 = {}; v8f acc11 = {};

#pragma unroll 2
    for (int kt = 0; kt < KTILES; ++kt) {
        const size_t b0c = (size_t)kt * NPADW + c0;
        const size_t b1c = (size_t)kt * NPADW + c1;
        const size_t a0c = (size_t)kt * MROWS + r0;
        const size_t a1c = (size_t)kt * MROWS + r1;
        const v16bf b0h = ldfrag(Whi, b0c, half);
        const v16bf b0l = ldfrag(Wlo, b0c, half);
        const v16bf b1h = ldfrag(Whi, b1c, half);
        const v16bf b1l = ldfrag(Wlo, b1c, half);
        const v16bf a0h = ldfrag(Ahi, a0c, half);
        const v16bf a0l = ldfrag(Alo, a0c, half);
        const v16bf a1h = ldfrag(Ahi, a1c, half);
        const v16bf a1l = ldfrag(Alo, a1c, half);
        acc00 = wmma_bf16(a0h, b0h, acc00);
        acc10 = wmma_bf16(a1h, b0h, acc10);
        acc01 = wmma_bf16(a0h, b1h, acc01);
        acc11 = wmma_bf16(a1h, b1h, acc11);
        acc00 = wmma_bf16(a0h, b0l, acc00);
        acc10 = wmma_bf16(a1h, b0l, acc10);
        acc01 = wmma_bf16(a0h, b1l, acc01);
        acc11 = wmma_bf16(a1h, b1l, acc11);
        acc00 = wmma_bf16(a0l, b0h, acc00);
        acc10 = wmma_bf16(a1l, b0h, acc10);
        acc01 = wmma_bf16(a0l, b1h, acc01);
        acc11 = wmma_bf16(a1l, b1h, acc11);
    }

    /* scaled logits -> LDS; padded columns masked to -inf */
    const int  cl0 = wn * 32 + lrow;
    const int  cl1 = cl0 + 16;
    const bool v0  = (blockCol + cl0) < NCLS;
    const bool v1  = (blockCol + cl1) < NCLS;
#pragma unroll
    for (int r = 0; r < 8; ++r) {
        const int m0 = wm * 32 + r + 8 * half;
        tile[m0][cl0]      = v0 ? SCALE * acc00[r] : -__builtin_inff();
        tile[m0][cl1]      = v1 ? SCALE * acc01[r] : -__builtin_inff();
        tile[m0 + 16][cl0] = v0 ? SCALE * acc10[r] : -__builtin_inff();
        tile[m0 + 16][cl1] = v1 ? SCALE * acc11[r] : -__builtin_inff();
    }
    __syncthreads();

    if (threadIdx.x < 64) {
        const int r    = threadIdx.x;
        const int grow = blockRow + r;
        float m = -__builtin_inff();
        for (int c = 0; c < 128; ++c) m = fmaxf(m, tile[r][c]);
        float s = 0.0f;
        for (int c = 0; c < 128; ++c) s += expf(tile[r][c] - m);
        float* pp = partials + ((size_t)grow * NBLK + blockIdx.x) * 2;
        pp[0] = m; pp[1] = s;
        const int lbl = labels[grow & (BATCH - 1)];
        if (lbl >= blockCol && lbl < blockCol + 128)
            lblLogit[grow] = tile[r][lbl - blockCol];
    }
}

/* -------- kernel 5: sim = v @ t^T + fused soft-margin pair loss --------
   grid (8, 16); B operand = normalized t rows = rows [1024,2048) of A bufs. */
__global__ __launch_bounds__(256) void sim_gemm_kernel(
    const __bf16* __restrict__ Ahi, const __bf16* __restrict__ Alo,
    const int* __restrict__ labels, float* __restrict__ simPart) {

    __shared__ float tile[64][132];
    __shared__ float red[256];

    const int lane = threadIdx.x & 31;
    const int wave = threadIdx.x >> 5;
    const int wm   = wave >> 2;
    const int wn   = wave & 3;
    const int half = lane >> 4;
    const int lrow = lane & 15;
    const int blockRow = blockIdx.y * 64;
    const int blockCol = blockIdx.x * 128;

    const int b0 = BATCH + blockCol + wn * 32 + lrow;  /* t rows as B columns */
    const int b1 = b0 + 16;
    const int r0 = blockRow + wm * 32 + lrow;
    const int r1 = r0 + 16;

    v8f acc00 = {}; v8f acc01 = {}; v8f acc10 = {}; v8f acc11 = {};

#pragma unroll 2
    for (int kt = 0; kt < KTILES; ++kt) {
        const size_t b0c = (size_t)kt * MROWS + b0;
        const size_t b1c = (size_t)kt * MROWS + b1;
        const size_t a0c = (size_t)kt * MROWS + r0;
        const size_t a1c = (size_t)kt * MROWS + r1;
        const v16bf b0h = ldfrag(Ahi, b0c, half);
        const v16bf b0l = ldfrag(Alo, b0c, half);
        const v16bf b1h = ldfrag(Ahi, b1c, half);
        const v16bf b1l = ldfrag(Alo, b1c, half);
        const v16bf a0h = ldfrag(Ahi, a0c, half);
        const v16bf a0l = ldfrag(Alo, a0c, half);
        const v16bf a1h = ldfrag(Ahi, a1c, half);
        const v16bf a1l = ldfrag(Alo, a1c, half);
        acc00 = wmma_bf16(a0h, b0h, acc00);
        acc10 = wmma_bf16(a1h, b0h, acc10);
        acc01 = wmma_bf16(a0h, b1h, acc01);
        acc11 = wmma_bf16(a1h, b1h, acc11);
        acc00 = wmma_bf16(a0h, b0l, acc00);
        acc10 = wmma_bf16(a1h, b0l, acc10);
        acc01 = wmma_bf16(a0h, b1l, acc01);
        acc11 = wmma_bf16(a1h, b1l, acc11);
        acc00 = wmma_bf16(a0l, b0h, acc00);
        acc10 = wmma_bf16(a1l, b0h, acc10);
        acc01 = wmma_bf16(a0l, b1h, acc01);
        acc11 = wmma_bf16(a1l, b1h, acc11);
    }

    const int cl0 = wn * 32 + lrow;
    const int cl1 = cl0 + 16;
#pragma unroll
    for (int r = 0; r < 8; ++r) {
        const int m0 = wm * 32 + r + 8 * half;
        tile[m0][cl0]      = acc00[r];
        tile[m0][cl1]      = acc01[r];
        tile[m0 + 16][cl0] = acc10[r];
        tile[m0 + 16][cl1] = acc11[r];
    }
    __syncthreads();

    float sum = 0.0f;
    for (int idx = threadIdx.x; idx < 64 * 128; idx += 256) {
        const int r = idx >> 7, c = idx & 127;
        const float s  = tile[r][c];
        const int   li = labels[blockRow + r];
        const int   lj = labels[blockCol + c];
        sum += (li == lj) ? softplusf(-SPOS * (s - ALPHA))
                          : softplusf( SNEG * (s - BETA));
    }
    red[threadIdx.x] = sum; __syncthreads();
    for (int off = 128; off > 0; off >>= 1) {
        if (threadIdx.x < off) red[threadIdx.x] += red[threadIdx.x + off];
        __syncthreads();
    }
    if (threadIdx.x == 0) simPart[blockIdx.y * gridDim.x + blockIdx.x] = red[0];
}

/* -------- kernel 6: per-row logsumexp merge of N-block partials -------- */
__global__ __launch_bounds__(256) void rowlse_kernel(const float* __restrict__ partials,
                                                     const float* __restrict__ lblLogit,
                                                     float* __restrict__ rowLoss) {
    const int r = blockIdx.x * 256 + threadIdx.x;
    if (r >= MROWS) return;
    const float* pp = partials + (size_t)r * NBLK * 2;
    float m = -__builtin_inff();
    for (int i = 0; i < NBLK; ++i) m = fmaxf(m, pp[2 * i]);
    float s = 0.0f;
    for (int i = 0; i < NBLK; ++i) s += pp[2 * i + 1] * expf(pp[2 * i] - m);
    rowLoss[r] = m + logf(s) - lblLogit[r];
}

/* -------- kernel 7: deterministic final sums -> out[0..1] -------- */
__global__ __launch_bounds__(256) void finalize_kernel(const float* __restrict__ rowLoss,
                                                       const float* __restrict__ simPart,
                                                       float* __restrict__ out) {
    __shared__ float red[256];
    float s = 0.0f;
    for (int i = threadIdx.x; i < MROWS; i += 256) s += rowLoss[i];
    red[threadIdx.x] = s; __syncthreads();
    for (int off = 128; off > 0; off >>= 1) {
        if (threadIdx.x < off) red[threadIdx.x] += red[threadIdx.x + off];
        __syncthreads();
    }
    if (threadIdx.x == 0) out[0] = red[0] / (float)BATCH;
    __syncthreads();

    red[threadIdx.x] = (threadIdx.x < 128) ? simPart[threadIdx.x] : 0.0f;
    __syncthreads();
    for (int off = 128; off > 0; off >>= 1) {
        if (threadIdx.x < off) red[threadIdx.x] += red[threadIdx.x + off];
        __syncthreads();
    }
    if (threadIdx.x == 0) out[1] = 2.0f * red[0] / (float)BATCH;
}

extern "C" void kernel_launch(void* const* d_in, const int* in_sizes, int n_in,
                              void* d_out, int out_size, void* d_ws, size_t ws_size,
                              hipStream_t stream) {
    const float* vis    = (const float*)d_in[0];
    const float* txt    = (const float*)d_in[1];
    const int*   labels = (const int*)d_in[2];
    const float* W      = (const float*)d_in[3];
    float*       out    = (float*)d_out;

    char* ws = (char*)d_ws;
    float*  rowInv   = (float*)(ws + B_ROWINV);
    float*  lblLogit = (float*)(ws + B_LBL);
    float*  partials = (float*)(ws + B_PART);
    float*  rowLoss  = (float*)(ws + B_ROWLOSS);
    float*  simPart  = (float*)(ws + B_SIMPART);
    __bf16* Ahi      = (__bf16*)(ws + B_AHI);
    __bf16* Alo      = (__bf16*)(ws + B_ALO);
    __bf16* Whi      = (__bf16*)(ws + B_WHI);
    __bf16* Wlo      = (__bf16*)(ws + B_WLO);

    rownorm_kernel<<<MROWS, 256, 0, stream>>>(vis, txt, rowInv);
    aconv_kernel<<<(MROWS * KTILES) / 256, 256, 0, stream>>>(vis, txt, rowInv, Ahi, Alo);
    wconv_kernel<<<NPADW / 256, 256, 0, stream>>>(W, Whi, Wlo);
    inst_gemm_kernel<<<dim3(NBLK, MROWS / 64), 256, 0, stream>>>(
        Ahi, Alo, Whi, Wlo, labels, partials, lblLogit);
    sim_gemm_kernel<<<dim3(BATCH / 128, BATCH / 64), 256, 0, stream>>>(
        Ahi, Alo, labels, simPart);
    rowlse_kernel<<<MROWS / 256, 256, 0, stream>>>(partials, lblLogit, rowLoss);
    finalize_kernel<<<1, 256, 0, stream>>>(rowLoss, simPart, out);
}
